// AudioGRUModel_18605798326639
// MI455X (gfx1250) — compile-verified, matrix operand order checked by reading, not run on hardware
//
#include <hip/hip_runtime.h>

// GRU: BATCH=256, INPUT=512, HIDDEN=512, STEPS=256
#define B_   256
#define IN_  512
#define H_   512
#define S_   256
#define G3_  1536   // 3*H
#define NT_  96     // total 16-wide col tiles in 3H
#define MT_  4096   // total 16-high row tiles in S*B

typedef __attribute__((ext_vector_type(8)))  __bf16 v8bf;
typedef __attribute__((ext_vector_type(16))) __bf16 v16bf;
typedef __attribute__((ext_vector_type(8)))  float  v8f;

union V16 { v16bf v; v8bf h[2]; };

// Build a 16-element bf16 WMMA operand from two contiguous 16B chunks at
// +0 and +16 elements (CDNA5 16-bit A/B per-lane layout: V0..3 = K+0..7,
// V4..7 = K+16..23 for this lane's K base).
__device__ __forceinline__ v16bf load16g(const __bf16* p) {
    V16 u;
    u.h[0] = *(const v8bf*)(p);
    u.h[1] = *(const v8bf*)(p + 16);
    return u.v;
}

// ---------------------------------------------------------------------------
// Init: zero h0 (ping buffer) and the global-barrier counter.
// ---------------------------------------------------------------------------
__global__ void init_state(__bf16* __restrict__ hb, unsigned int* __restrict__ cnt) {
    int i = blockIdx.x * 256 + threadIdx.x;
    if (i < B_ * H_) hb[i] = (__bf16)0.0f;
    if (i == 0) *cnt = 0u;
}

// ---------------------------------------------------------------------------
// Convert W_ih [1536x512] and W_hh [1536x512] fp32 -> bf16 (row-major; a
// WMMA-B lane for column n reads row n of W, K-contiguous).
// ---------------------------------------------------------------------------
__global__ void w_convert(const float* __restrict__ wih, const float* __restrict__ whh,
                          __bf16* __restrict__ wihb, __bf16* __restrict__ whhb) {
    int i = blockIdx.x * 256 + threadIdx.x;
    const int N = G3_ * IN_;
    if (i < N)  wihb[i] = (__bf16)wih[i];
    else        whhb[i - N] = (__bf16)whh[i - N];
}

// ---------------------------------------------------------------------------
// Transpose x[b][i][s] fp32 -> xT[(s*B + b)][i] bf16 via LDS 32x32 tiles so
// reads coalesce along s and writes coalesce along i.
// ---------------------------------------------------------------------------
__global__ __launch_bounds__(256)
void x_transpose(const float* __restrict__ x, __bf16* __restrict__ xT) {
    __shared__ float tile[32][33];
    const int tx = threadIdx.x, ty = threadIdx.y;         // (32, 8)
    const int ib = blockIdx.x * 32;                       // input-dim tile
    const int sb = blockIdx.y * 32;                       // step tile
    const int b  = blockIdx.z;
#pragma unroll
    for (int r = 0; r < 4; ++r)
        tile[ty + 8 * r][tx] = x[((size_t)b * IN_ + ib + ty + 8 * r) * S_ + sb + tx];
    __syncthreads();
#pragma unroll
    for (int r = 0; r < 4; ++r) {
        int s = sb + ty + 8 * r;
        xT[((size_t)s * B_ + b) * IN_ + ib + tx] = (__bf16)tile[tx][ty + 8 * r];
    }
}

// ---------------------------------------------------------------------------
// Big input-projection GEMM: gi = xT[S*B,512] * W_ih^T (bf16 WMMA, f32 accum).
// gi is stored in WMMA C-FRAGMENT TILE ORDER:
//   gi[((mt*96 + ntile)*32 + lane)*8 + r]  (v8bf per lane per tile)
// so each lane's 8 accumulators become ONE global_store_b128 (512B contiguous
// per wave-tile) and the recurrent consumer reloads them with ONE
// global_load_b128 per gate. Each wave: 16(M)x64(N), A reused over 4 N-tiles;
// the block's W_ih strip is staged once into LDS in per-lane layout (64 KB).
// Grid: 24 strips x 512 blocks, 8 waves/block.
// ---------------------------------------------------------------------------
__global__ __launch_bounds__(256)
void gru_gi_gemm(const __bf16* __restrict__ xT, const __bf16* __restrict__ wih,
                 __bf16* __restrict__ gi) {
    __shared__ v8bf ldsB[4 * 16 * 32 * 2];   // 4 n-subtiles x 16 kb x 32 lanes x 32B = 64 KB
    const int tid  = threadIdx.x;
    const int lane = tid & 31;
    const int wid  = tid >> 5;
    const int strip = blockIdx.x >> 9;                 // same for all waves in block
    const int mt    = ((blockIdx.x & 511) << 3) + wid; // 0..4095
    const int l15   = lane & 15;
    const int kofs  = (lane & 16) ? 8 : 0;

    // stage B tiles: 64 tiles, 8 per wave
#pragma unroll
    for (int q = 0; q < 8; ++q) {
        int tt = wid * 8 + q;               // 0..63
        int sub = tt >> 4, kb = tt & 15;
        const __bf16* src = wih + (size_t)(strip * 64 + sub * 16 + l15) * IN_ + kb * 32 + kofs;
        int idx = (tt * 32 + lane) * 2;
        ldsB[idx]     = *(const v8bf*)(src);
        ldsB[idx + 1] = *(const v8bf*)(src + 16);
    }
    __syncthreads();

    v8f z = {};
    v8f acc[4] = { z, z, z, z };
    const __bf16* arow = xT + (size_t)(mt * 16 + l15) * IN_ + kofs;
#pragma unroll 4
    for (int kb = 0; kb < 16; ++kb) {
        v16bf a = load16g(arow + kb * 32);
#pragma unroll
        for (int sub = 0; sub < 4; ++sub) {
            V16 u;
            int idx = (((sub << 4) + kb) * 32 + lane) * 2;
            u.h[0] = ldsB[idx];
            u.h[1] = ldsB[idx + 1];
            acc[sub] = __builtin_amdgcn_wmma_f32_16x16x32_bf16(
                false, a, false, u.v, (short)0, acc[sub], false, false);
        }
    }
    // Packed C-fragment store: one b128 per n-subtile.
#pragma unroll
    for (int sub = 0; sub < 4; ++sub) {
        int ntile = strip * 4 + sub;
        v8bf pk;
#pragma unroll
        for (int r = 0; r < 8; ++r) pk[r] = (__bf16)acc[sub][r];
        *(v8bf*)(gi + (((size_t)mt * NT_ + ntile) * 32 + lane) * 8) = pk;
    }
}

// ---------------------------------------------------------------------------
// Persistent recurrent kernel: 64 blocks x 8 waves = 512 waves, one 16x16
// hidden tile each (16 batch-tiles x 32 col-tiles). Per step each wave runs
// three K=512 WMMA chains (r/z/n gates; A from h reused across gates, B from
// time-invariant W_hh held in LDS), reads its three packed gi fragments with
// b128 loads (prefetched one step ahead), applies the GRU nonlinearity, and
// ping-pongs h in bf16. Steps separated by a software grid barrier
// (atomic arrive + acquire spin + s_sleep).
// ---------------------------------------------------------------------------
#define REC_BLOCKS 64

__global__ __launch_bounds__(256)
void gru_recurrent(const __bf16* __restrict__ gi, const __bf16* __restrict__ whh,
                   const float* __restrict__ bih, const float* __restrict__ bhh,
                   __bf16* __restrict__ hb, float* __restrict__ out,
                   unsigned int* __restrict__ cnt) {
    __shared__ v8bf ldsB[3 * 16 * 32 * 2];   // 3 gates x 16 kb x 32 lanes x 32B = 48 KB
    const int tid  = threadIdx.x;
    const int lane = tid & 31;
    const int wid  = tid >> 5;
    const int wg   = blockIdx.x * 8 + wid;   // 0..511
    const int tile_m = wg & 15;              // batch tile (waves 8k..8k+7 share tile_n)
    const int tile_n = wg >> 4;              // hidden-col tile 0..31
    const int l15  = lane & 15;
    const int kofs = (lane & 16) ? 8 : 0;
    const int rofs = (lane & 16) ? 8 : 0;
    const int col  = tile_n * 16 + l15;

    // Stage time-invariant W_hh B-tiles for this block's column tile: 48 tiles, 6/wave.
#pragma unroll
    for (int q = 0; q < 6; ++q) {
        int tt = wid * 6 + q;                // 0..47
        int g = tt >> 4, kb = tt & 15;
        const __bf16* src = whh + (size_t)(g * H_ + tile_n * 16 + l15) * H_ + kb * 32 + kofs;
        int idx = (tt * 32 + lane) * 2;
        ldsB[idx]     = *(const v8bf*)(src);
        ldsB[idx + 1] = *(const v8bf*)(src + 16);
    }
    __syncthreads();

    const float bi0 = bih[col],          bh0 = bhh[col];
    const float bi1 = bih[H_ + col],     bh1 = bhh[H_ + col];
    const float bi2 = bih[2 * H_ + col], bh2 = bhh[2 * H_ + col];

    // gi fragment addresses for this wave: gate g -> ntile = g*32 + tile_n.
    // Per-step row-tile mt = t*16 + tile_m.
    const size_t lane8   = (size_t)lane * 8;
    const size_t g0base  = ((size_t)(0 * 32 + tile_n)) * 32 * 8 + lane8;
    const size_t g1base  = ((size_t)(1 * 32 + tile_n)) * 32 * 8 + lane8;
    const size_t g2base  = ((size_t)(2 * 32 + tile_n)) * 32 * 8 + lane8;
    const size_t mtstep  = (size_t)NT_ * 32 * 8;     // elements per row-tile

    for (int t = 0; t < S_; ++t) {
        const __bf16* hcur = hb + (size_t)(t & 1) * (B_ * H_);
        __bf16*       hnxt = hb + (size_t)((t + 1) & 1) * (B_ * H_);
        const size_t  mtb  = (size_t)(t * 16 + tile_m) * mtstep;

        // Prefetch next step's gi fragments while this step's WMMAs run.
        if (t + 1 < S_) {
            const __bf16* nxt = gi + mtb + 16 * mtstep;
            __builtin_prefetch(nxt + g0base, 0, 3);
            __builtin_prefetch(nxt + g1base, 0, 3);
            __builtin_prefetch(nxt + g2base, 0, 3);
        }

        v8f accr = {}, accz = {}, accn = {};
        const __bf16* arow = hcur + (size_t)(tile_m * 16 + l15) * H_ + kofs;
#pragma unroll 4
        for (int kb = 0; kb < 16; ++kb) {
            v16bf a = load16g(arow + kb * 32);
            V16 u0, u1, u2;
            int i0 = ((0 * 16 + kb) * 32 + lane) * 2;
            int i1 = ((1 * 16 + kb) * 32 + lane) * 2;
            int i2 = ((2 * 16 + kb) * 32 + lane) * 2;
            u0.h[0] = ldsB[i0]; u0.h[1] = ldsB[i0 + 1];
            u1.h[0] = ldsB[i1]; u1.h[1] = ldsB[i1 + 1];
            u2.h[0] = ldsB[i2]; u2.h[1] = ldsB[i2 + 1];
            accr = __builtin_amdgcn_wmma_f32_16x16x32_bf16(false, a, false, u0.v, (short)0, accr, false, false);
            accz = __builtin_amdgcn_wmma_f32_16x16x32_bf16(false, a, false, u1.v, (short)0, accz, false, false);
            accn = __builtin_amdgcn_wmma_f32_16x16x32_bf16(false, a, false, u2.v, (short)0, accn, false, false);
        }

        // Packed gi fragment loads: one b128 per gate.
        v8bf gr = *(const v8bf*)(gi + mtb + g0base);
        v8bf gz = *(const v8bf*)(gi + mtb + g1base);
        v8bf gn = *(const v8bf*)(gi + mtb + g2base);

        // Gate epilogue: h_new = (1-z)*tanh(i_n + r*h_n) + z*h
#pragma unroll
        for (int r = 0; r < 8; ++r) {
            int m = tile_m * 16 + r + rofs;
            float ir = (float)gr[r] + bi0;
            float iz = (float)gz[r] + bi1;
            float in = (float)gn[r] + bi2;
            float hr = accr[r] + bh0;
            float hz = accz[r] + bh1;
            float hn = accn[r] + bh2;
            float rg = 1.0f / (1.0f + __expf(-(ir + hr)));
            float zg = 1.0f / (1.0f + __expf(-(iz + hz)));
            float e  = __expf(2.0f * (in + rg * hn));
            float ng = (e - 1.0f) / (e + 1.0f);
            float ho = (float)hcur[(size_t)m * H_ + col];
            float hv = (1.0f - zg) * ng + zg * ho;
            hnxt[(size_t)m * H_ + col] = (__bf16)hv;
            if (t == S_ - 1) out[(size_t)m * H_ + col] = hv;
        }

        // Grid-wide barrier (monotonic counter: no reset race).
        __threadfence();
        __syncthreads();
        if (tid == 0) {
            __hip_atomic_fetch_add(cnt, 1u, __ATOMIC_ACQ_REL, __HIP_MEMORY_SCOPE_AGENT);
            unsigned target = (unsigned)(t + 1) * (unsigned)REC_BLOCKS;
            while (__hip_atomic_load(cnt, __ATOMIC_ACQUIRE, __HIP_MEMORY_SCOPE_AGENT) < target)
                __builtin_amdgcn_s_sleep(2);
        }
        __syncthreads();
    }
}

// ---------------------------------------------------------------------------
// Launch
// ---------------------------------------------------------------------------
extern "C" void kernel_launch(void* const* d_in, const int* in_sizes, int n_in,
                              void* d_out, int out_size, void* d_ws, size_t ws_size,
                              hipStream_t stream) {
    const float* x   = (const float*)d_in[0];   // [B, IN, S]
    const float* wih = (const float*)d_in[1];   // [3H, IN]
    const float* whh = (const float*)d_in[2];   // [3H, H]
    const float* bih = (const float*)d_in[3];   // [3H]
    const float* bhh = (const float*)d_in[4];   // [3H]
    float* out = (float*)d_out;                 // [B, H] fp32

    // Workspace carve-out (~272 MB):
    char* ws = (char*)d_ws;
    __bf16* xT   = (__bf16*)ws;  ws += (size_t)S_ * B_ * IN_ * 2;   //  67 MB
    __bf16* wihb = (__bf16*)ws;  ws += (size_t)G3_ * IN_ * 2;       // 1.5 MB
    __bf16* whhb = (__bf16*)ws;  ws += (size_t)G3_ * H_ * 2;        // 1.5 MB
    __bf16* gi   = (__bf16*)ws;  ws += (size_t)MT_ * NT_ * 32 * 8 * 2; // 201 MB (tile order)
    __bf16* hb   = (__bf16*)ws;  ws += (size_t)2 * B_ * H_ * 2;     // 0.5 MB
    unsigned int* cnt = (unsigned int*)ws;

    init_state<<<512, 256, 0, stream>>>(hb, cnt);
    w_convert<<<(2 * G3_ * IN_) / 256, 256, 0, stream>>>(wih, whh, wihb, whhb);
    x_transpose<<<dim3(IN_ / 32, S_ / 32, B_), dim3(32, 8), 0, stream>>>(x, xT);
    gru_gi_gemm<<<24 * 512, 256, 0, stream>>>(xT, wihb, gi);
    gru_recurrent<<<REC_BLOCKS, 256, 0, stream>>>(gi, whhb, bih, bhh, hb, out, cnt);
}
